// FlamingoMaskedCrossAttention_14130442404428
// MI455X (gfx1250) — compile-verified, matrix operand order checked by reading, not run on hardware
//
#include <hip/hip_runtime.h>

// ---------------- types ----------------
typedef __attribute__((ext_vector_type(16))) __bf16 v16bf;
typedef __attribute__((ext_vector_type(8)))  __bf16 v8bf;
typedef __attribute__((ext_vector_type(8)))  float  v8f;
typedef __attribute__((ext_vector_type(4)))  int    v4i;

#define NB    4      // batch
#define TT    2048   // text tokens
#define DMODEL 2048
#define DVIS  1024
#define NHEAD 16
#define DHEAD 64
#define INNER 1024   // NHEAD*DHEAD
#define JTOT  512    // T_IMG * N_LAT
#define NWIN  8      // T_IMG
#define LN_EPS 1e-5f

// GEMM tiling
#define BM 128
#define BN 128
#define BK 32
#define LDP 40       // LDS row pitch (bf16 elems): 80B stride -> conflict-free frag reads

#if __has_builtin(__builtin_amdgcn_global_load_async_to_lds_b128)
#define ASYNC_LDS 1
#else
#define ASYNC_LDS 0
#endif

static __device__ __forceinline__ unsigned short f2bf(float f) {
  unsigned int u = __float_as_uint(f);
  unsigned int r = (u + 0x7FFFu + ((u >> 16) & 1u)) >> 16;  // RNE
  return (unsigned short)r;
}

// Load a 16x32 bf16 WMMA fragment (A layout; B^T uses identical addressing).
// lanes 0-15 -> row=lane, K = {0..7, 16..23}; lanes 16-31 -> row=lane-16, K = {8..15, 24..31}.
static __device__ __forceinline__ v16bf load_frag(const unsigned short* base, int ld,
                                                  int row, int kbase) {
  const __bf16* p = (const __bf16*)base + (size_t)row * ld + kbase;
  v8bf lo = *(const v8bf*)(p);
  v8bf hi = *(const v8bf*)(p + 16);
  v16bf f;
#pragma unroll
  for (int i = 0; i < 8; ++i) { f[i] = lo[i]; f[i + 8] = hi[i]; }
  return f;
}

static __device__ __forceinline__ v8f wmma_bf16(v16bf a, v16bf b, v8f c) {
  return __builtin_amdgcn_wmma_f32_16x16x32_bf16(false, a, false, b, (short)0, c,
                                                 false, false);
}

#if ASYNC_LDS
typedef __attribute__((address_space(1))) v4i* gptr_v4i;
typedef __attribute__((address_space(3))) v4i* lptr_v4i;

// async copy 16B global -> LDS (per-lane addresses), tracked by ASYNCcnt
static __device__ __forceinline__ void async_cp16(const void* g, void* l) {
  __builtin_amdgcn_global_load_async_to_lds_b128((gptr_v4i)(g), (lptr_v4i)(l), 0, 0);
}
#endif

static __device__ __forceinline__ void wait_async0() {
#if ASYNC_LDS
#if __has_builtin(__builtin_amdgcn_s_wait_asynccnt)
  __builtin_amdgcn_s_wait_asynccnt(0);
#else
  asm volatile("s_wait_asynccnt 0" ::: "memory");
#endif
#endif
}

// ---------------- tiny prep kernels ----------------

__global__ void tt_kernel(const unsigned char* __restrict__ ml, int* __restrict__ tt) {
  int b = blockIdx.x;
  if (threadIdx.x == 0) {
    int c = 0;
    for (int t = 0; t < TT; ++t) {
      c += (ml[b * TT + t] != 0) ? 1 : 0;
      tt[b * TT + t] = c;
    }
  }
}

__global__ void cvt_kernel(const float* __restrict__ src, unsigned short* __restrict__ dst,
                           int n) {
  int i = blockIdx.x * 256 + threadIdx.x;
  if (i < n) dst[i] = f2bf(src[i]);
}

// W (K x N, f32) -> W^T (N x K, bf16)
__global__ void wt_kernel(const float* __restrict__ W, unsigned short* __restrict__ WT,
                          int K, int N) {
  int idx = blockIdx.x * 256 + threadIdx.x;
  if (idx < N * K) {
    int n = idx / K, k = idx % K;
    WT[idx] = f2bf(W[(size_t)k * N + n]);
  }
}

__global__ __launch_bounds__(256) void ln_kernel(const float* __restrict__ x,
                                                 const float* __restrict__ w,
                                                 const float* __restrict__ bias,
                                                 unsigned short* __restrict__ xn) {
  __shared__ float red[256];
  int row = blockIdx.x;
  const float* xr = x + (size_t)row * DMODEL;
  float s = 0.f, s2 = 0.f;
  for (int j = threadIdx.x; j < DMODEL; j += 256) {
    float v = xr[j];
    s += v; s2 += v * v;
  }
  red[threadIdx.x] = s; __syncthreads();
  for (int o = 128; o > 0; o >>= 1) {
    if (threadIdx.x < o) red[threadIdx.x] += red[threadIdx.x + o];
    __syncthreads();
  }
  float mean = red[0] * (1.f / DMODEL); __syncthreads();
  red[threadIdx.x] = s2; __syncthreads();
  for (int o = 128; o > 0; o >>= 1) {
    if (threadIdx.x < o) red[threadIdx.x] += red[threadIdx.x + o];
    __syncthreads();
  }
  float var = red[0] * (1.f / DMODEL) - mean * mean;
  float inv = rsqrtf(var + LN_EPS);
  for (int j = threadIdx.x; j < DMODEL; j += 256) {
    float v = (xr[j] - mean) * inv * w[j] + bias[j];
    xn[(size_t)row * DMODEL + j] = f2bf(v);
  }
}

// ---------------- LDS-tiled 8-wave WMMA GEMM: C = alpha * A(MxK) * BT(NxK)^T ----------------
// Block tile 128x128, K-slice 32, double-buffered LDS. Waves: 4 along M x 2 along N,
// each wave owns a 32x64 register tile (2x4 accumulators).
// MODE 0: f32 out.  MODE 1: bf16 out.  MODE 2: kv split (K normal, V transposed per head).

struct TileRegs { v8bf a[2]; v8bf b[2]; };

static __device__ __forceinline__ void load_tile_regs(const unsigned short* A,
                                                      const unsigned short* BT, int K,
                                                      int m_blk, int n_blk, int k0,
                                                      int tid, TileRegs& t) {
#pragma unroll
  for (int c = 0; c < 2; ++c) {
    int idx = c * 256 + tid;            // 512 chunks of 16B per tile
    int row = idx >> 2, chk = idx & 3;  // 4 chunks per 32-elem row
    t.a[c] = *(const v8bf*)((const __bf16*)A + (size_t)(m_blk + row) * K + k0 + chk * 8);
    t.b[c] = *(const v8bf*)((const __bf16*)BT + (size_t)(n_blk + row) * K + k0 + chk * 8);
  }
}

static __device__ __forceinline__ void store_tile_lds(unsigned short* sA,
                                                      unsigned short* sB, int tid,
                                                      const TileRegs& t) {
#pragma unroll
  for (int c = 0; c < 2; ++c) {
    int idx = c * 256 + tid;
    int row = idx >> 2, chk = idx & 3;
    *(v8bf*)((__bf16*)sA + row * LDP + chk * 8) = t.a[c];
    *(v8bf*)((__bf16*)sB + row * LDP + chk * 8) = t.b[c];
  }
}

#if ASYNC_LDS
static __device__ __forceinline__ void async_tile(const unsigned short* A,
                                                  const unsigned short* BT, int K,
                                                  int m_blk, int n_blk, int k0, int tid,
                                                  unsigned short* sA, unsigned short* sB) {
#pragma unroll
  for (int c = 0; c < 2; ++c) {
    int idx = c * 256 + tid;
    int row = idx >> 2, chk = idx & 3;
    async_cp16((const __bf16*)A + (size_t)(m_blk + row) * K + k0 + chk * 8,
               (__bf16*)sA + row * LDP + chk * 8);
    async_cp16((const __bf16*)BT + (size_t)(n_blk + row) * K + k0 + chk * 8,
               (__bf16*)sB + row * LDP + chk * 8);
  }
}
#endif

static __device__ __forceinline__ void compute_tile(const unsigned short* sA,
                                                    const unsigned short* sB, int wm,
                                                    int wn, int rsel, int khalf,
                                                    v8f (&acc)[2][4]) {
  v16bf a0 = load_frag(sA, LDP, wm * 32 + rsel, khalf);
  v16bf a1 = load_frag(sA, LDP, wm * 32 + 16 + rsel, khalf);
#pragma unroll
  for (int ns = 0; ns < 4; ++ns) {
    v16bf b = load_frag(sB, LDP, wn * 64 + ns * 16 + rsel, khalf);
    acc[0][ns] = wmma_bf16(a0, b, acc[0][ns]);
    acc[1][ns] = wmma_bf16(a1, b, acc[1][ns]);
  }
}

template <int MODE>
__global__ __launch_bounds__(256) void gemm_tiled(const unsigned short* __restrict__ A,
                                                  const unsigned short* __restrict__ BT,
                                                  void* __restrict__ out0,
                                                  void* __restrict__ out1, int M, int N,
                                                  int K, float alpha) {
  __shared__ __align__(16) unsigned short sA[2][BM * LDP];
  __shared__ __align__(16) unsigned short sB[2][BN * LDP];

  const int tid = threadIdx.x;
  const int lane = tid & 31, wid = tid >> 5;
  const int wm = wid & 3, wn = wid >> 2;  // 4 x 2 wave grid
  const int rsel = lane & 15;
  const int khalf = (lane >> 4) << 3;
  const int mhalf = (lane >> 4) * 8;
  const int m_blk = blockIdx.y * BM, n_blk = blockIdx.x * BN;

  v8f acc[2][4];
#pragma unroll
  for (int ms = 0; ms < 2; ++ms)
#pragma unroll
    for (int ns = 0; ns < 4; ++ns)
#pragma unroll
      for (int r = 0; r < 8; ++r) acc[ms][ns][r] = 0.f;

  const int nk = K / BK;
#if ASYNC_LDS
  async_tile(A, BT, K, m_blk, n_blk, 0, tid, sA[0], sB[0]);
  wait_async0();
  __syncthreads();
  for (int kc = 0; kc < nk; ++kc) {
    int cur = kc & 1;
    if (kc + 1 < nk)
      async_tile(A, BT, K, m_blk, n_blk, (kc + 1) * BK, tid, sA[cur ^ 1], sB[cur ^ 1]);
    compute_tile(sA[cur], sB[cur], wm, wn, rsel, khalf, acc);
    if (kc + 1 < nk) wait_async0();
    __syncthreads();
  }
#else
  TileRegs tr;
  load_tile_regs(A, BT, K, m_blk, n_blk, 0, tid, tr);
  store_tile_lds(sA[0], sB[0], tid, tr);
  __syncthreads();
  for (int kc = 0; kc < nk; ++kc) {
    int cur = kc & 1;
    if (kc + 1 < nk) load_tile_regs(A, BT, K, m_blk, n_blk, (kc + 1) * BK, tid, tr);
    compute_tile(sA[cur], sB[cur], wm, wn, rsel, khalf, acc);
    if (kc + 1 < nk) store_tile_lds(sA[cur ^ 1], sB[cur ^ 1], tid, tr);
    __syncthreads();
  }
#endif

  const int m0 = m_blk + wm * 32, n0 = n_blk + wn * 64;
#pragma unroll
  for (int ms = 0; ms < 2; ++ms)
#pragma unroll
    for (int ns = 0; ns < 4; ++ns)
#pragma unroll
      for (int r = 0; r < 8; ++r) {
        int m = m0 + ms * 16 + mhalf + r;
        int n = n0 + ns * 16 + rsel;
        float v = acc[ms][ns][r] * alpha;
        if (MODE == 0) {
          ((float*)out0)[(size_t)m * N + n] = v;
        } else if (MODE == 1) {
          ((unsigned short*)out0)[(size_t)m * N + n] = f2bf(v);
        } else {  // kv projection: N==2*INNER, M==NB*JTOT
          if (n < INNER) {
            ((unsigned short*)out0)[(size_t)m * INNER + n] = f2bf(v);
          } else {
            int h = (n - INNER) >> 6, d = (n - INNER) & 63;
            int bb = m >> 9, j = m & 511;
            ((unsigned short*)out1)[(size_t)((bb * NHEAD + h) * DHEAD + d) * JTOT + j] =
                f2bf(v);
          }
        }
      }
}

// ---------------- fused masked attention ----------------
__global__ __launch_bounds__(32) void attn_kernel(const unsigned short* __restrict__ qb,
                                                  const unsigned short* __restrict__ kb,
                                                  const unsigned short* __restrict__ vt,
                                                  const int* __restrict__ text_time,
                                                  unsigned short* __restrict__ ao) {
  __shared__ float sS[32 * 64];
  __shared__ __align__(16) unsigned short sP[32 * 64];

  int b = blockIdx.z, h = blockIdx.y, i0 = blockIdx.x * 32;
  int lane = threadIdx.x;
  int rsel = lane & 15;
  int khalf = (lane >> 4) << 3;
  int mhalf = (lane >> 4) * 8;

  v16bf aq[2][2];
#pragma unroll
  for (int ms = 0; ms < 2; ++ms)
#pragma unroll
    for (int ks = 0; ks < 2; ++ks)
      aq[ms][ks] = load_frag(qb, INNER, b * TT + i0 + ms * 16 + rsel,
                             h * DHEAD + ks * 32 + khalf);

  v8f o[8];
#pragma unroll
  for (int t = 0; t < 8; ++t)
#pragma unroll
    for (int r = 0; r < 8; ++r) o[t][r] = 0.f;

  int tt = text_time[b * TT + i0 + lane];
  const unsigned short* vbase = vt + (size_t)((b * NHEAD + h) * DHEAD) * JTOT;

  for (int w = 0; w < NWIN; ++w) {
    int j0 = w * 64;
    v8f s[8];
#pragma unroll
    for (int t = 0; t < 8; ++t)
#pragma unroll
      for (int r = 0; r < 8; ++r) s[t][r] = 0.f;
#pragma unroll
    for (int ks = 0; ks < 2; ++ks) {
#pragma unroll
      for (int ns = 0; ns < 4; ++ns) {
        v16bf bk = load_frag(kb, INNER, b * JTOT + j0 + ns * 16 + rsel,
                             h * DHEAD + ks * 32 + khalf);
        s[0 * 4 + ns] = wmma_bf16(aq[0][ks], bk, s[0 * 4 + ns]);
        s[1 * 4 + ns] = wmma_bf16(aq[1][ks], bk, s[1 * 4 + ns]);
      }
    }
#pragma unroll
    for (int t = 0; t < 8; ++t) {
      int ms = t >> 2, ns = t & 3;
#pragma unroll
      for (int r = 0; r < 8; ++r)
        sS[(ms * 16 + mhalf + r) * 64 + ns * 16 + rsel] = s[t][r];
    }
    __syncthreads();
    {
      float* row = &sS[lane * 64];
      unsigned short* prow = &sP[lane * 64];
      if (tt == w + 1) {
        float mx = row[0];
#pragma unroll 8
        for (int j = 1; j < 64; ++j) mx = fmaxf(mx, row[j]);
        float sum = 0.f;
#pragma unroll 8
        for (int j = 0; j < 64; ++j) {
          float e = __expf(row[j] - mx);
          row[j] = e;
          sum += e;
        }
        float inv = 1.f / sum;
#pragma unroll 8
        for (int j = 0; j < 64; ++j) prow[j] = f2bf(row[j] * inv);
      } else {
#pragma unroll 8
        for (int j = 0; j < 64; ++j) prow[j] = 0;  // covers tt==0 (no media)
      }
    }
    __syncthreads();
#pragma unroll
    for (int ks = 0; ks < 2; ++ks) {
      v16bf ap0 = load_frag(sP, 64, rsel, ks * 32 + khalf);
      v16bf ap1 = load_frag(sP, 64, 16 + rsel, ks * 32 + khalf);
#pragma unroll
      for (int ns = 0; ns < 4; ++ns) {
        v16bf bv = load_frag(vbase, JTOT, ns * 16 + rsel, j0 + ks * 32 + khalf);
        o[0 * 4 + ns] = wmma_bf16(ap0, bv, o[0 * 4 + ns]);
        o[1 * 4 + ns] = wmma_bf16(ap1, bv, o[1 * 4 + ns]);
      }
    }
    __syncthreads();
  }
#pragma unroll
  for (int t = 0; t < 8; ++t) {
    int ms = t >> 2, ns = t & 3;
#pragma unroll
    for (int r = 0; r < 8; ++r) {
      int m = i0 + ms * 16 + mhalf + r;
      int col = h * DHEAD + ns * 16 + rsel;
      ao[(size_t)(b * TT + m) * INNER + col] = f2bf(o[t][r]);
    }
  }
}

// ---------------- launch ----------------
extern "C" void kernel_launch(void* const* d_in, const int* in_sizes, int n_in,
                              void* d_out, int out_size, void* d_ws, size_t ws_size,
                              hipStream_t stream) {
  const float* x      = (const float*)d_in[0];
  const float* media  = (const float*)d_in[1];
  const float* norm_w = (const float*)d_in[2];
  const float* norm_b = (const float*)d_in[3];
  const float* Wq     = (const float*)d_in[4];
  const float* Wkv    = (const float*)d_in[5];
  const float* Wo     = (const float*)d_in[6];
  const unsigned char* ml = (const unsigned char*)d_in[7];

  char* ws = (char*)d_ws;
  size_t off = 0;
  auto carve = [&](size_t bytes) {
    off = (off + 255) & ~(size_t)255;
    char* p = ws + off;
    off += bytes;
    return p;
  };
  int* tt             = (int*)carve((size_t)NB * TT * 4);
  unsigned short* xn  = (unsigned short*)carve((size_t)NB * TT * DMODEL * 2);
  unsigned short* mb  = (unsigned short*)carve((size_t)NB * JTOT * DVIS * 2);
  unsigned short* WqT = (unsigned short*)carve((size_t)INNER * DMODEL * 2);
  unsigned short* WkvT= (unsigned short*)carve((size_t)2 * INNER * DVIS * 2);
  unsigned short* WoT = (unsigned short*)carve((size_t)DMODEL * INNER * 2);
  unsigned short* qb  = (unsigned short*)carve((size_t)NB * TT * INNER * 2);
  unsigned short* kbuf= (unsigned short*)carve((size_t)NB * JTOT * INNER * 2);
  unsigned short* vtb = (unsigned short*)carve((size_t)NB * NHEAD * DHEAD * JTOT * 2);
  unsigned short* ao  = (unsigned short*)carve((size_t)NB * TT * INNER * 2);
  (void)ws_size;

  tt_kernel<<<NB, 32, 0, stream>>>(ml, tt);
  ln_kernel<<<NB * TT, 256, 0, stream>>>(x, norm_w, norm_b, xn);
  {
    int n = NB * JTOT * DVIS;
    cvt_kernel<<<(n + 255) / 256, 256, 0, stream>>>(media, mb, n);
  }
  wt_kernel<<<(INNER * DMODEL + 255) / 256, 256, 0, stream>>>(Wq, WqT, DMODEL, INNER);
  wt_kernel<<<(2 * INNER * DVIS + 255) / 256, 256, 0, stream>>>(Wkv, WkvT, DVIS, 2 * INNER);
  wt_kernel<<<(DMODEL * INNER + 255) / 256, 256, 0, stream>>>(Wo, WoT, INNER, DMODEL);

  // Q = (xn @ Wq) * 1/8  -> bf16
  gemm_tiled<1><<<dim3(INNER / BN, NB * TT / BM), 256, 0, stream>>>(
      xn, WqT, qb, nullptr, NB * TT, INNER, DMODEL, 0.125f);
  // KV = media @ Wkv -> K (row-major), V (transposed per head)
  gemm_tiled<2><<<dim3(2 * INNER / BN, NB * JTOT / BM), 256, 0, stream>>>(
      mb, WkvT, kbuf, vtb, NB * JTOT, 2 * INNER, DVIS, 1.0f);
  // fused masked attention
  attn_kernel<<<dim3(TT / 32, NHEAD, NB), 32, 0, stream>>>(qb, kbuf, vtb, tt, ao);
  // out = ao @ Wo -> f32
  gemm_tiled<0><<<dim3(DMODEL / BN, NB * TT / BM), 256, 0, stream>>>(
      ao, WoT, (float*)d_out, nullptr, NB * TT, DMODEL, INNER, 1.0f);
}
